// SparseSEBlock_17849884082697
// MI455X (gfx1250) — matrix operand order repaired and checked by reading, not compiled, not run on hardware
//
#include <hip/hip_runtime.h>
#include <math.h>

typedef __attribute__((ext_vector_type(4)))  float    f4;
typedef __attribute__((ext_vector_type(8)))  float    v8f;
typedef __attribute__((ext_vector_type(16))) _Float16 v16h;

#define CC   256   // channels
#define BB   32    // batch size
#define HH   64    // hidden (CC / 4)
#define POOL_ROWS 512

// ---------------------------------------------------------------------------
// Zero the pooled-sum + counts accumulators (harness does not re-poison ws).
// ---------------------------------------------------------------------------
__global__ void zero_init_kernel(float* __restrict__ p, int n) {
  for (int i = threadIdx.x; i < n; i += blockDim.x) p[i] = 0.0f;
}

// ---------------------------------------------------------------------------
// Segment-sum pooling. bidx is sorted, so each thread run-length-accumulates
// and flushes to global atomics only on segment change (~2 flushes / block).
// 256 threads = 4 row-streams x 64 float4 column groups -> b128 coalesced.
// ---------------------------------------------------------------------------
__global__ void pool_kernel(const float* __restrict__ feats,
                            const int*   __restrict__ bidx,
                            float* __restrict__ pooled_sum,
                            float* __restrict__ counts,
                            int n) {
  const int tid    = threadIdx.x;
  const int strm   = tid >> 6;   // 0..3 row stream
  const int c4     = tid & 63;   // float4 column group
  const f4* feats4 = (const f4*)feats;

  long long blockStart = (long long)blockIdx.x * POOL_ROWS;
  long long rend = blockStart + POOL_ROWS;
  if (rend > (long long)n) rend = (long long)n;

  f4  acc = {0.f, 0.f, 0.f, 0.f};
  int cur = -1;
  int runlen = 0;

  for (long long r = blockStart + strm; r < rend; r += 4) {
    int s = bidx[r];   // uniform across wave -> scalar broadcast load
    if (s != cur) {
      if (cur >= 0) {
        float* dst = pooled_sum + (long long)cur * CC + c4 * 4;
        atomicAdd(dst + 0, acc.x);
        atomicAdd(dst + 1, acc.y);
        atomicAdd(dst + 2, acc.z);
        atomicAdd(dst + 3, acc.w);
        if (c4 == 0) atomicAdd(&counts[cur], (float)runlen);
      }
      cur = s;
      acc = (f4){0.f, 0.f, 0.f, 0.f};
      runlen = 0;
    }
    f4 v = feats4[r * (CC / 4) + c4];
    acc += v;
    ++runlen;
  }
  if (cur >= 0) {
    float* dst = pooled_sum + (long long)cur * CC + c4 * 4;
    atomicAdd(dst + 0, acc.x);
    atomicAdd(dst + 1, acc.y);
    atomicAdd(dst + 2, acc.z);
    atomicAdd(dst + 3, acc.w);
    if (c4 == 0) atomicAdd(&counts[cur], (float)runlen);
  }
}

// ---------------------------------------------------------------------------
// Tiny SE MLP on a single wave (32 threads, EXEC all-1s) using CDNA5 WMMA
// v_wmma_f32_16x16x32_f16.  h = relu(pooled @ fc1_w^T + b1)  [32 x 64]
//                           se = sigmoid(h @ fc2_w^T + b2)   [32 x 256]
// Fragment layouts per CDNA5 ISA:
//   A (16x32 f16, MxK): lane l -> M = l&15, element j -> K = (j&7) + 16*(j>>3) + 8*(l>>4)
//   B (32x16 f16, KxN): lane l -> N = l&15, element j -> K = (j&7) + 16*(j>>3) + 8*(l>>4)
//   C/D (16x16 f32):    element r -> M = r + 8*(l>>4), N = l&15
// ---------------------------------------------------------------------------
__global__ void se_mlp_kernel(const float* __restrict__ pooled_sum,
                              const float* __restrict__ counts,
                              const float* __restrict__ fc1_w,  // [64][256]
                              const float* __restrict__ fc1_b,  // [64]
                              const float* __restrict__ fc2_w,  // [256][64]
                              const float* __restrict__ fc2_b,  // [256]
                              float* __restrict__ se) {         // [32][256]
  __shared__ _Float16 Ap[BB][CC];   // pooled means, f16
  __shared__ _Float16 H[BB][HH];    // hidden activations, f16

  const int l    = threadIdx.x;  // 0..31
  const int half = l >> 4;       // 0 or 1
  const int m    = l & 15;

  // pooled mean -> LDS (f16)
  for (int i = l; i < BB * CC; i += 32) {
    int row = i >> 8;
    float cnt = counts[row];
    if (cnt < 1.0f) cnt = 1.0f;
    Ap[row][i & (CC - 1)] = (_Float16)(pooled_sum[i] / cnt);
  }
  __syncthreads();

  // ---- fc1: [32x256] @ [256x64] -> relu -> H ----
  for (int mt = 0; mt < 2; ++mt) {
    for (int nt = 0; nt < HH / 16; ++nt) {
      float bias = fc1_b[nt * 16 + m];
      v8f acc;
      for (int r = 0; r < 8; ++r) acc[r] = bias;
      for (int k0 = 0; k0 < CC; k0 += 32) {
        v16h a, b;
        for (int j = 0; j < 16; ++j) {
          a[j] = Ap[mt * 16 + m][k0 + ((j >> 3) << 4) + (half << 3) + (j & 7)];
          b[j] = (_Float16)fc1_w[(nt * 16 + m) * CC + k0 + ((j >> 3) << 4) + (half << 3) + (j & 7)];
        }
        acc = __builtin_amdgcn_wmma_f32_16x16x32_f16(
            false, a, false, b, (short)0, acc, false, false);
      }
      for (int r = 0; r < 8; ++r) {
        float v = acc[r];
        if (v < 0.0f) v = 0.0f;   // relu
        H[mt * 16 + r + (half << 3)][nt * 16 + m] = (_Float16)v;
      }
    }
  }
  __syncthreads();

  // ---- fc2: [32x64] @ [64x256] -> sigmoid -> se ----
  for (int mt = 0; mt < 2; ++mt) {
    for (int nt = 0; nt < CC / 16; ++nt) {
      float bias = fc2_b[nt * 16 + m];
      v8f acc;
      for (int r = 0; r < 8; ++r) acc[r] = bias;
      for (int k0 = 0; k0 < HH; k0 += 32) {
        v16h a, b;
        for (int j = 0; j < 16; ++j) {
          a[j] = H[mt * 16 + m][k0 + ((j >> 3) << 4) + (half << 3) + (j & 7)];
          b[j] = (_Float16)fc2_w[(nt * 16 + m) * HH + k0 + ((j >> 3) << 4) + (half << 3) + (j & 7)];
        }
        acc = __builtin_amdgcn_wmma_f32_16x16x32_f16(
            false, a, false, b, (short)0, acc, false, false);
      }
      for (int r = 0; r < 8; ++r) {
        float v = acc[r];
        se[(mt * 16 + r + (half << 3)) * CC + nt * 16 + m] =
            1.0f / (1.0f + expf(-v));
      }
    }
  }
}

// ---------------------------------------------------------------------------
// Reweight: out = feats * se[bidx].  Blocks process the array in REVERSE so
// the tail of feats that pooling just pulled through the 192 MB L2 is hit
// first (saves ~190 MB of HBM re-reads).  NT stores: output is never reused.
// ---------------------------------------------------------------------------
__global__ void reweight_kernel(const f4* __restrict__ feats4,
                                const int* __restrict__ bidx,
                                const f4* __restrict__ se4,   // [32][64]
                                f4* __restrict__ out4,
                                long long total4) {
  long long cb  = (long long)(gridDim.x - 1 - blockIdx.x);
  long long idx = cb * blockDim.x + threadIdx.x;
  if (idx >= total4) return;
  long long row = idx >> 6;
  int c4 = (int)(idx & 63);
  int s  = bidx[row];                 // uniform across half-wave -> broadcast
  f4 g   = se4[s * (CC / 4) + c4];    // 32 KB table, L1/L2 resident
  f4 v   = feats4[idx];
  v *= g;
  __builtin_nontemporal_store(v, &out4[idx]);
}

// ---------------------------------------------------------------------------
extern "C" void kernel_launch(void* const* d_in, const int* in_sizes, int n_in,
                              void* d_out, int out_size, void* d_ws, size_t ws_size,
                              hipStream_t stream) {
  const float* feats = (const float*)d_in[0];
  const float* fc1_w = (const float*)d_in[1];
  const float* fc1_b = (const float*)d_in[2];
  const float* fc2_w = (const float*)d_in[3];
  const float* fc2_b = (const float*)d_in[4];
  const int*   bidx  = (const int*)d_in[5];
  // d_in[6] = batch_size scalar (fixed at 32 by the reference)

  const int n = in_sizes[0] / CC;  // number of feature rows

  float* ws         = (float*)d_ws;
  float* pooled_sum = ws;                 // [32][256]
  float* counts     = ws + BB * CC;       // [32]
  float* se         = ws + BB * CC + BB;  // [32][256] (16B aligned: offset 8224)

  // 1) zero accumulators
  zero_init_kernel<<<1, 256, 0, stream>>>(pooled_sum, BB * CC + BB);

  // 2) segment-sum pooling
  int pgrid = (n + POOL_ROWS - 1) / POOL_ROWS;
  pool_kernel<<<pgrid, 256, 0, stream>>>(feats, bidx, pooled_sum, counts, n);

  // 3) SE MLP on one wave (WMMA)
  se_mlp_kernel<<<1, 32, 0, stream>>>(pooled_sum, counts,
                                      fc1_w, fc1_b, fc2_w, fc2_b, se);

  // 4) reweight (reverse order for L2 reuse)
  long long total4 = (long long)n * (CC / 4);
  int rgrid = (int)((total4 + 255) / 256);
  reweight_kernel<<<rgrid, 256, 0, stream>>>((const f4*)feats, bidx,
                                             (const f4*)se, (f4*)d_out, total4);
}
